// AttnReadout_8306466751032
// MI455X (gfx1250) — compile-verified, matrix operand order checked by reading, not run on hardware
//
#include <hip/hip_runtime.h>
#include <hip/hip_bf16.h>
#include <math.h>

// ---------------------------------------------------------------------------
// AttnReadout for MI455X (gfx1250, wave32, WMMA + async LDS staging).
// Pipeline: BN-stats -> BN-fold -> feat_v WMMA GEMM -> score WMMA GEMM ->
//           segment bounds -> per-graph softmax + weighted readout.
// ---------------------------------------------------------------------------

typedef __attribute__((ext_vector_type(16))) _Float16 v16h;
typedef __attribute__((ext_vector_type(8)))  _Float16 v8h;
typedef __attribute__((ext_vector_type(8)))  float    v8f;

#define DIMS 128
#define WPB  8   // waves per block in WMMA kernels (256 threads)

// ---------------------------------------------------------------------------
// Stage a 128x128 f32 weight matrix into LDS via CDNA5 async global->LDS DMA
// (GLOBAL_LOAD_ASYNC_TO_LDS_B128, tracked by ASYNCcnt). Each lane moves 16B
// per issue; 256 threads x 16 iterations covers 64KB with no VGPR round-trip.
// ---------------------------------------------------------------------------
__device__ __forceinline__ void stage_weights_async(const float* __restrict__ Wsrc,
                                                    float* Wlds) {
  for (int i = threadIdx.x * 4; i < DIMS * DIMS; i += 256 * 4) {
    unsigned ldsoff = (unsigned)(unsigned long long)(Wlds + i);  // LDS byte offset
    const float* gp = Wsrc + i;
    asm volatile("global_load_async_to_lds_b128 %0, %1, off"
                 :
                 : "v"(ldsoff), "v"(gp)
                 : "memory");
  }
  asm volatile("s_wait_asynccnt 0x0" ::: "memory");
}

// Load 8 consecutive normalized features (x = feat*scale + shift) as f16.
__device__ __forceinline__ v8h norm8_to_f16(const float* __restrict__ rowp,
                                            const float* __restrict__ sc,
                                            const float* __restrict__ sh,
                                            int k) {
  float4 f0 = *reinterpret_cast<const float4*>(rowp + k);
  float4 f1 = *reinterpret_cast<const float4*>(rowp + k + 4);
  float4 s0 = *reinterpret_cast<const float4*>(sc + k);
  float4 s1 = *reinterpret_cast<const float4*>(sc + k + 4);
  float4 h0 = *reinterpret_cast<const float4*>(sh + k);
  float4 h1 = *reinterpret_cast<const float4*>(sh + k + 4);
  v8h r;
  r[0] = (_Float16)fmaf(f0.x, s0.x, h0.x);
  r[1] = (_Float16)fmaf(f0.y, s0.y, h0.y);
  r[2] = (_Float16)fmaf(f0.z, s0.z, h0.z);
  r[3] = (_Float16)fmaf(f0.w, s0.w, h0.w);
  r[4] = (_Float16)fmaf(f1.x, s1.x, h1.x);
  r[5] = (_Float16)fmaf(f1.y, s1.y, h1.y);
  r[6] = (_Float16)fmaf(f1.z, s1.z, h1.z);
  r[7] = (_Float16)fmaf(f1.w, s1.w, h1.w);
  return r;
}

// Build A fragment for 16-bit A 16x32 layout:
// lane half (kbase = 0 or 8): elems 0..7 = K kbase..+7, elems 8..15 = K kbase+16..+23
__device__ __forceinline__ v16h buildA(const float* __restrict__ rowp,
                                       const float* __restrict__ sc,
                                       const float* __restrict__ sh,
                                       int t, int kbase) {
  v8h lo = norm8_to_f16(rowp, sc, sh, t * 32 + kbase);
  v8h hi = norm8_to_f16(rowp, sc, sh, t * 32 + kbase + 16);
  v16h a;
#pragma unroll
  for (int i = 0; i < 8; ++i) { a[i] = lo[i]; a[8 + i] = hi[i]; }
  return a;
}

// ---------------------------------------------------------------------------
// Kernel 1: per-feature sum / sum-of-squares (BatchNorm stats)
// ---------------------------------------------------------------------------
__global__ __launch_bounds__(256)
void bn_stats_kernel(const float* __restrict__ feat, float* __restrict__ sums,
                     float* __restrict__ sumsq, int N) {
  const int d    = threadIdx.x & (DIMS - 1);
  const int part = threadIdx.x >> 7;   // 0 or 1: two rows in flight per block
  float s = 0.f, q = 0.f;
  for (int row = blockIdx.x * 2 + part; row < N; row += gridDim.x * 2) {
    float v = feat[(size_t)row * DIMS + d];
    s += v;
    q += v * v;
  }
  __shared__ float ls[256], lq[256];
  ls[threadIdx.x] = s; lq[threadIdx.x] = q;
  __syncthreads();
  if (part == 0) {
    atomicAdd(&sums[d],  ls[d] + ls[d + DIMS]);
    atomicAdd(&sumsq[d], lq[d] + lq[d + DIMS]);
  }
}

// ---------------------------------------------------------------------------
// Kernel 2: fold BN into scale/shift
// ---------------------------------------------------------------------------
__global__ __launch_bounds__(128)
void bn_finalize_kernel(const float* __restrict__ sums, const float* __restrict__ sumsq,
                        const float* __restrict__ gamma, const float* __restrict__ beta,
                        float* __restrict__ scale, float* __restrict__ shift, int N) {
  const int d = threadIdx.x;
  const float invN = 1.0f / (float)N;
  float mean = sums[d] * invN;
  float var  = fmaxf(sumsq[d] * invN - mean * mean, 0.0f);
  float rstd = rsqrtf(var + 1e-5f);
  float sc   = gamma[d] * rstd;
  scale[d] = sc;
  shift[d] = beta[d] - mean * sc;
}

// ---------------------------------------------------------------------------
// Kernel 3: feat_v = x[last_nodes] @ W_v^T + b_v   (WMMA, gathered rows)
// One wave per 16-row tile; W_v async-staged to LDS, converted to f16 once.
// ---------------------------------------------------------------------------
__global__ __launch_bounds__(256)
void featv_wmma_kernel(const float* __restrict__ feat, const float* __restrict__ scale,
                       const float* __restrict__ shift, const float* __restrict__ Wv,
                       const float* __restrict__ bv, const int* __restrict__ lastn,
                       float* __restrict__ featv, int numTiles, int Mrows) {
  __shared__ float    Wst[DIMS * DIMS];  // 64KB async-staged f32
  __shared__ _Float16 Wl[DIMS * DIMS];   // 32KB f16 for WMMA B fragments
  stage_weights_async(Wv, Wst);
  __syncthreads();
  for (int i = threadIdx.x; i < DIMS * DIMS; i += 256) Wl[i] = (_Float16)Wst[i];
  __syncthreads();

  const int wave = threadIdx.x >> 5;
  const int lane = threadIdx.x & 31;
  const int lid  = lane & 15;
  const int halfsel = lane >> 4;
  const int kbase = halfsel * 8;
  const int tile = blockIdx.x * WPB + wave;
  if (tile >= numTiles) return;

  const int gr   = min(tile * 16 + lid, Mrows - 1);
  const int srow = lastn[gr];
  const float* rowp = feat + (size_t)srow * DIMS;

  v16h A[4];
#pragma unroll
  for (int t = 0; t < 4; ++t) A[t] = buildA(rowp, scale, shift, t, kbase);

#pragma unroll
  for (int ht = 0; ht < 8; ++ht) {
    const int col = ht * 16 + lid;
    v8f acc = {};
#pragma unroll
    for (int t = 0; t < 4; ++t) {
      const v8h blo = *reinterpret_cast<const v8h*>(Wl + col * DIMS + t * 32 + kbase);
      const v8h bhi = *reinterpret_cast<const v8h*>(Wl + col * DIMS + t * 32 + kbase + 16);
      v16h Bf;
#pragma unroll
      for (int i = 0; i < 8; ++i) { Bf[i] = blo[i]; Bf[8 + i] = bhi[i]; }
      acc = __builtin_amdgcn_wmma_f32_16x16x32_f16(false, A[t], false, Bf,
                                                   (short)0, acc, false, false);
    }
    const float bias = bv[col];
#pragma unroll
    for (int r = 0; r < 8; ++r) {
      int orow = tile * 16 + (halfsel ? r + 8 : r);
      if (orow < Mrows) featv[(size_t)orow * DIMS + col] = acc[r] + bias;
    }
  }
}

// ---------------------------------------------------------------------------
// Kernel 4: e[n] = dot(sigmoid(x@W_u^T + feat_v[seg]), w_e)   (WMMA)
// ---------------------------------------------------------------------------
__global__ __launch_bounds__(256)
void score_wmma_kernel(const float* __restrict__ feat, const float* __restrict__ scale,
                       const float* __restrict__ shift, const float* __restrict__ Wu,
                       const float* __restrict__ featv, const float* __restrict__ we,
                       const int* __restrict__ seg, float* __restrict__ eout,
                       int numTiles, int Mrows) {
  __shared__ float    Wst[DIMS * DIMS];  // 64KB async-staged f32
  __shared__ _Float16 Wl[DIMS * DIMS];   // 32KB f16 for WMMA B fragments
  stage_weights_async(Wu, Wst);
  __syncthreads();
  for (int i = threadIdx.x; i < DIMS * DIMS; i += 256) Wl[i] = (_Float16)Wst[i];
  __syncthreads();

  const int wave = threadIdx.x >> 5;
  const int lane = threadIdx.x & 31;
  const int lid  = lane & 15;
  const int halfsel = lane >> 4;
  const int kbase = halfsel * 8;
  const int tile = blockIdx.x * WPB + wave;
  if (tile >= numTiles) return;

  const int rowA = min(tile * 16 + lid, Mrows - 1);
  const float* rowp = feat + (size_t)rowA * DIMS;

  v16h A[4];
#pragma unroll
  for (int t = 0; t < 4; ++t) A[t] = buildA(rowp, scale, shift, t, kbase);

  int segids[8];
#pragma unroll
  for (int r = 0; r < 8; ++r) {
    int gr = tile * 16 + (halfsel ? r + 8 : r);
    segids[r] = seg[min(gr, Mrows - 1)];
  }

  float eacc[8];
#pragma unroll
  for (int r = 0; r < 8; ++r) eacc[r] = 0.f;

#pragma unroll
  for (int ht = 0; ht < 8; ++ht) {
    const int col = ht * 16 + lid;
    v8f acc = {};
#pragma unroll
    for (int t = 0; t < 4; ++t) {
      const v8h blo = *reinterpret_cast<const v8h*>(Wl + col * DIMS + t * 32 + kbase);
      const v8h bhi = *reinterpret_cast<const v8h*>(Wl + col * DIMS + t * 32 + kbase + 16);
      v16h Bf;
#pragma unroll
      for (int i = 0; i < 8; ++i) { Bf[i] = blo[i]; Bf[8 + i] = bhi[i]; }
      acc = __builtin_amdgcn_wmma_f32_16x16x32_f16(false, A[t], false, Bf,
                                                   (short)0, acc, false, false);
    }
    const float wcol = we[col];
#pragma unroll
    for (int r = 0; r < 8; ++r) {
      float u = acc[r] + featv[(size_t)segids[r] * DIMS + col];
      float sgm = 1.0f / (1.0f + __expf(-u));
      eacc[r] += sgm * wcol;
    }
  }

  // Reduce each row's partial score across the 16 lanes of its half-wave.
#pragma unroll
  for (int r = 0; r < 8; ++r) {
    float v = eacc[r];
    v += __shfl_xor(v, 1, 32);
    v += __shfl_xor(v, 2, 32);
    v += __shfl_xor(v, 4, 32);
    v += __shfl_xor(v, 8, 32);
    if (lid == 0) {
      int gr = tile * 16 + (halfsel ? r + 8 : r);
      if (gr < Mrows) eout[gr] = v;
    }
  }
}

// ---------------------------------------------------------------------------
// Kernel 5: per-graph [start,end) via binary search on sorted segment_ids
// ---------------------------------------------------------------------------
__global__ __launch_bounds__(256)
void bounds_kernel(const int* __restrict__ seg, int N, int B,
                   int* __restrict__ startArr, int* __restrict__ endArr) {
  int b = blockIdx.x * blockDim.x + threadIdx.x;
  if (b >= B) return;
  int lo = 0, hi = N;
  while (lo < hi) { int mid = (lo + hi) >> 1; if (seg[mid] < b) lo = mid + 1; else hi = mid; }
  int s = lo;
  hi = N;
  while (lo < hi) { int mid = (lo + hi) >> 1; if (seg[mid] < b + 1) lo = mid + 1; else hi = mid; }
  startArr[b] = s;
  endArr[b]   = lo;
}

// ---------------------------------------------------------------------------
// Kernel 6: per-graph softmax over e + alpha-weighted readout of x
// One 128-thread block per graph; thread = feature.
// ---------------------------------------------------------------------------
__global__ __launch_bounds__(128)
void readout_kernel(const float* __restrict__ feat, const float* __restrict__ scale,
                    const float* __restrict__ shift, const float* __restrict__ e,
                    const int* __restrict__ startArr, const int* __restrict__ endArr,
                    float* __restrict__ out) {
  const int b = blockIdx.x;
  const int d = threadIdx.x;
  const int s = startArr[b];
  const int t = endArr[b];

  __shared__ float red[DIMS];

  // max over segment
  float m = -INFINITY;
  for (int n = s + d; n < t; n += DIMS) m = fmaxf(m, e[n]);
  red[d] = m;
  __syncthreads();
  for (int off = 64; off > 0; off >>= 1) {
    if (d < off) red[d] = fmaxf(red[d], red[d + off]);
    __syncthreads();
  }
  m = red[0];
  __syncthreads();

  // sum of exp
  float sum = 0.f;
  for (int n = s + d; n < t; n += DIMS) sum += __expf(e[n] - m);
  red[d] = sum;
  __syncthreads();
  for (int off = 64; off > 0; off >>= 1) {
    if (d < off) red[d] += red[d + off];
    __syncthreads();
  }
  const float denom  = red[0];
  const float invden = (t > s) ? 1.0f / denom : 0.0f;

  // weighted readout, renormalizing feat on the fly; prefetch rows ahead
  const float sc = scale[d], sh = shift[d];
  float acc = 0.f;
  for (int n = s; n < t; ++n) {
    int np = n + 4 < t ? n + 4 : n;
    __builtin_prefetch(&feat[(size_t)np * DIMS + d], 0, 1);
    float w = __expf(e[n] - m) * invden;
    acc = fmaf(w, fmaf(feat[(size_t)n * DIMS + d], sc, sh), acc);
  }
  out[(size_t)b * DIMS + d] = acc;
}

// ---------------------------------------------------------------------------
// Launcher
// ---------------------------------------------------------------------------
extern "C" void kernel_launch(void* const* d_in, const int* in_sizes, int n_in,
                              void* d_out, int out_size, void* d_ws, size_t ws_size,
                              hipStream_t stream) {
  const float* feat  = (const float*)d_in[0];
  const float* gamma = (const float*)d_in[1];
  const float* beta  = (const float*)d_in[2];
  const float* Wu    = (const float*)d_in[3];
  const float* Wv    = (const float*)d_in[4];
  const float* bv    = (const float*)d_in[5];
  const float* we    = (const float*)d_in[6];
  const int*   seg   = (const int*)d_in[7];
  const int*   lastn = (const int*)d_in[8];
  const int N = in_sizes[7];   // number of nodes
  const int B = in_sizes[8];   // number of graphs

  // Workspace layout (floats unless noted):
  float* ws     = (float*)d_ws;
  float* sums   = ws;                 // [128]
  float* sumsq  = ws + 128;           // [128]
  float* scale  = ws + 256;           // [128]
  float* shift  = ws + 384;           // [128]
  float* featv  = ws + 512;           // [B,128]
  float* evec   = featv + (size_t)B * DIMS;  // [N]
  int*   startA = (int*)(evec + (size_t)N);  // [B]
  int*   endA   = startA + B;                // [B]

  hipMemsetAsync(sums, 0, 256 * sizeof(float), stream);

  bn_stats_kernel<<<2048, 256, 0, stream>>>(feat, sums, sumsq, N);
  bn_finalize_kernel<<<1, 128, 0, stream>>>(sums, sumsq, gamma, beta, scale, shift, N);

  const int featvTiles = (B + 15) / 16;
  featv_wmma_kernel<<<(featvTiles + WPB - 1) / WPB, 256, 0, stream>>>(
      feat, scale, shift, Wv, bv, lastn, featv, featvTiles, B);

  const int scoreTiles = (N + 15) / 16;
  score_wmma_kernel<<<(scoreTiles + WPB - 1) / WPB, 256, 0, stream>>>(
      feat, scale, shift, Wu, featv, we, seg, evec, scoreTiles, N);

  bounds_kernel<<<(B + 255) / 256, 256, 0, stream>>>(seg, N, B, startA, endA);

  readout_kernel<<<B, 128, 0, stream>>>(feat, scale, shift, evec, startA, endA,
                                        (float*)d_out);
}
